// CGCNNStyle_27127013441817
// MI455X (gfx1250) — matrix operand order repaired and checked by reading, not compiled, not run on hardware
//
#include <hip/hip_runtime.h>
#include <hip/hip_bf16.h>

#define NN   50000          // nodes
#define NG   100            // graphs
#define NPG  500            // nodes per graph
#define NE   800000         // edges
#define EMBD 128
#define NL   4
#define BN_EPS 1e-5f

typedef __attribute__((ext_vector_type(16))) __bf16 v16bf;
typedef __attribute__((ext_vector_type(8)))  float  v8f;

union FragV { v16bf v; uint4 q[2]; };

__device__ inline unsigned short f2bf(float f) {
    unsigned u = __float_as_uint(f);
    unsigned r = u + 0x7FFFu + ((u >> 16) & 1u);   // round-to-nearest-even
    return (unsigned short)(r >> 16);
}
__device__ inline float bf2f(unsigned short h) {
    return __uint_as_float(((unsigned)h) << 16);
}
__device__ inline float sigmoidf_(float x) { return 1.0f / (1.0f + __expf(-x)); }
__device__ inline float softplusf_(float x) { return (x > 20.0f) ? x : log1pf(__expf(x)); }

// ---------------------------------------------------------------------------
// Weight prep: Wcat[l][o][k] (bf16), o<128: Wf_top^T, 128..255: Wf_mid^T,
// 256..383: Ws_top^T, 384..511: Ws_mid^T.  (row o contiguous over k)
// ---------------------------------------------------------------------------
__global__ void prep_w_kernel(const float* __restrict__ Wf, const float* __restrict__ Ws,
                              unsigned short* __restrict__ Wcat) {
    int idx = blockIdx.x * 256 + threadIdx.x;        // [0, 4*512*128)
    if (idx >= NL * 512 * 128) return;
    int k = idx & 127;
    int o = (idx >> 7) & 511;
    int l = idx >> 16;                               // 512*128 == 65536
    const float* W = (o < 256) ? Wf : Ws;
    int oo = o & 127;
    int part = (o >> 7) & 1;                         // 0: rows 0..127, 1: rows 128..255
    float v = W[(size_t)l * 257 * 128 + (size_t)(part * 128 + k) * 128 + oo];
    Wcat[idx] = f2bf(v);
}

__global__ void embed_kernel(const int* __restrict__ z, const float* __restrict__ emb,
                             float* __restrict__ xf, unsigned short* __restrict__ xb) {
    int idx = blockIdx.x * 256 + threadIdx.x;        // [0, NN*128)
    if (idx >= NN * EMBD) return;
    int n = idx >> 7, c = idx & 127;
    float v = emb[(size_t)z[n] * EMBD + c];
    xf[idx] = v;
    xb[idx] = f2bf(v);
}

__global__ void dist_kernel(const int* __restrict__ ei, const float* __restrict__ pos,
                            float* __restrict__ dist) {
    int e = blockIdx.x * 256 + threadIdx.x;
    if (e >= NE) return;
    int j = ei[e];            // source
    int i = ei[NE + e];       // target
    float dx = pos[3 * i + 0] - pos[3 * j + 0];
    float dy = pos[3 * i + 1] - pos[3 * j + 1];
    float dz = pos[3 * i + 2] - pos[3 * j + 2];
    dist[e] = sqrtf(dx * dx + dy * dy + dz * dz);
}

// ---------------------------------------------------------------------------
// Node GEMM: P16[NN,512] (bf16) = x_bf16[NN,128] @ Wcat_l^T, f32 accumulate via
// v_wmma_f32_16x16x32_bf16. Wave computes one 16x16 tile, K in 4 steps of 32.
// Output kept in bf16 so the edge-phase gather set stays L2-resident (192MB).
// ---------------------------------------------------------------------------
__global__ void gemm_node_kernel(const unsigned short* __restrict__ xb,
                                 const unsigned short* __restrict__ Wl,  // [512][128] bf16
                                 unsigned short* __restrict__ P16) {
    int w    = threadIdx.x >> 5;
    int lane = threadIdx.x & 31;
    int m0 = blockIdx.x << 4;                        // 3125 tiles cover 50000 rows
    int n0 = ((blockIdx.y << 3) + w) << 4;           // 32 tiles cover 512 cols
    int ln   = lane & 15;
    int half = lane >> 4;

    v8f acc = {};
    const unsigned short* arow = xb + (size_t)(m0 + ln) * 128;
    const unsigned short* brow = Wl + (size_t)(n0 + ln) * 128;

#pragma unroll
    for (int k0 = 0; k0 < 128; k0 += 32) {
        FragV fa, fb;
        // A (16x32 bf16): lane row = ln; elems 0..7 -> K=half*8+0..7, 8..15 -> K=16+half*8+0..7
        fa.q[0] = *(const uint4*)(arow + k0 + half * 8);
        fa.q[1] = *(const uint4*)(arow + k0 + half * 8 + 16);
        // B (32x16 bf16): lane col = ln; elems 0..7 -> K=half*8+0..7, 8..15 -> K=16+half*8+0..7
        const unsigned short* bp = brow + k0 + half * 8;
        fb.q[0] = *(const uint4*)(bp);
        fb.q[1] = *(const uint4*)(bp + 16);
        acc = __builtin_amdgcn_wmma_f32_16x16x32_bf16(
            false, fa.v, false, fb.v, (short)0, acc, false, false);
    }
    // D (16x16 f32): lane col = ln; VGPR r -> row r + half*8; store bf16
#pragma unroll
    for (int r = 0; r < 8; ++r)
        P16[(size_t)(m0 + half * 8 + r) * 512 + (n0 + ln)] = f2bf(acc[r]);
}

__global__ void zero_aggstats_kernel(float* __restrict__ agg, float* __restrict__ stats) {
    int idx = blockIdx.x * 256 + threadIdx.x;
    if (idx < NN * EMBD) agg[idx] = 0.0f;
    else if (idx < NN * EMBD + 256) stats[idx - NN * EMBD] = 0.0f;
}

__global__ void zero_stats_kernel(float* __restrict__ stats) {
    if (threadIdx.x < 256) stats[threadIdx.x] = 0.0f;
}

// ---------------------------------------------------------------------------
// Edge phase: msg = sigmoid(Pf_i + Pf_j + d*wfl + bf) * softplus(Ps_i + Ps_j + d*wsl + bs)
// scatter-add into agg[i] (f32 atomics for accumulate fidelity).
// One wave per edge, 4 channels per lane; bf16 gathers = 1KB/edge, L2-resident.
// ---------------------------------------------------------------------------
__global__ void edge_msg_kernel(const int* __restrict__ ei, const float* __restrict__ dist,
                                const unsigned short* __restrict__ P16,
                                const float* __restrict__ Wf, const float* __restrict__ Ws,
                                const float* __restrict__ bfp, const float* __restrict__ bsp,
                                int layer, float* __restrict__ agg) {
    int t = blockIdx.x * 256 + threadIdx.x;
    int e = t >> 5;
    if (e >= NE) return;
    int lane = t & 31;
    int c = lane << 2;
    int j = ei[e];
    int i = ei[NE + e];
    float d = dist[e];

    const ushort4 afi = *(const ushort4*)(P16 + (size_t)i * 512 + c);
    const ushort4 bfj = *(const ushort4*)(P16 + (size_t)j * 512 + 128 + c);
    const ushort4 asi = *(const ushort4*)(P16 + (size_t)i * 512 + 256 + c);
    const ushort4 bsj = *(const ushort4*)(P16 + (size_t)j * 512 + 384 + c);
    const size_t lastrow = (size_t)layer * 257 * 128 + 256 * 128;
    const float4 wfl = *(const float4*)(Wf + lastrow + c);
    const float4 wsl = *(const float4*)(Ws + lastrow + c);
    const float4 bfv = *(const float4*)(bfp + layer * 128 + c);
    const float4 bsv = *(const float4*)(bsp + layer * 128 + c);

    float* ag = agg + (size_t)i * EMBD + c;
    float pf, ps;
    pf = bf2f(afi.x) + bf2f(bfj.x) + d * wfl.x + bfv.x;
    ps = bf2f(asi.x) + bf2f(bsj.x) + d * wsl.x + bsv.x;
    atomicAdd(ag + 0, sigmoidf_(pf) * softplusf_(ps));
    pf = bf2f(afi.y) + bf2f(bfj.y) + d * wfl.y + bfv.y;
    ps = bf2f(asi.y) + bf2f(bsj.y) + d * wsl.y + bsv.y;
    atomicAdd(ag + 1, sigmoidf_(pf) * softplusf_(ps));
    pf = bf2f(afi.z) + bf2f(bfj.z) + d * wfl.z + bfv.z;
    ps = bf2f(asi.z) + bf2f(bsj.z) + d * wsl.z + bsv.z;
    atomicAdd(ag + 2, sigmoidf_(pf) * softplusf_(ps));
    pf = bf2f(afi.w) + bf2f(bfj.w) + d * wfl.w + bfv.w;
    ps = bf2f(asi.w) + bf2f(bsj.w) + d * wsl.w + bsv.w;
    atomicAdd(ag + 3, sigmoidf_(pf) * softplusf_(ps));
}

// ---------------------------------------------------------------------------
// BatchNorm (biased var) over NN rows, 128 channels.
// ---------------------------------------------------------------------------
__global__ void bn_reduce_kernel(const float* __restrict__ src, float* __restrict__ stats) {
    int c = threadIdx.x;                             // blockDim = 128
    float s = 0.0f, s2 = 0.0f;
    for (int r = blockIdx.x; r < NN; r += gridDim.x) {
        float v = src[(size_t)r * EMBD + c];
        s += v; s2 += v * v;
    }
    atomicAdd(&stats[c], s);
    atomicAdd(&stats[128 + c], s2);
}

__global__ void bn_apply1_kernel(const float* __restrict__ agg, const float* __restrict__ stats,
                                 const float* __restrict__ g1, const float* __restrict__ b1c,
                                 const float* __restrict__ xf, int layer,
                                 float* __restrict__ outb) {
    int idx = blockIdx.x * 256 + threadIdx.x;
    if (idx >= NN * EMBD) return;
    int c = idx & 127;
    const float invN = 1.0f / (float)NN;
    float mu  = stats[c] * invN;
    float var = stats[128 + c] * invN - mu * mu;
    float nrm = (agg[idx] - mu) * rsqrtf(var + BN_EPS);
    outb[idx] = nrm * g1[layer * 128 + c] + b1c[layer * 128 + c] + xf[idx];
}

__global__ void bn_apply2_kernel(const float* __restrict__ outb, const float* __restrict__ stats,
                                 const float* __restrict__ g2, const float* __restrict__ b2c,
                                 int layer, float* __restrict__ xf, unsigned short* __restrict__ xb) {
    int idx = blockIdx.x * 256 + threadIdx.x;
    if (idx >= NN * EMBD) return;
    int c = idx & 127;
    const float invN = 1.0f / (float)NN;
    float mu  = stats[c] * invN;
    float var = stats[128 + c] * invN - mu * mu;
    float nrm = (outb[idx] - mu) * rsqrtf(var + BN_EPS);
    float v = fmaxf(nrm * g2[layer * 128 + c] + b2c[layer * 128 + c], 0.0f);
    xf[idx] = v;
    xb[idx] = f2bf(v);
}

// ---------------------------------------------------------------------------
// Mean pool (batch is contiguous blocks of NPG nodes) + MLP head.
// ---------------------------------------------------------------------------
__global__ void pool_kernel(const float* __restrict__ xf, float* __restrict__ g) {
    int gid = blockIdx.x, c = threadIdx.x;           // blockDim = 128
    float s = 0.0f;
    const float* base = xf + (size_t)gid * NPG * EMBD + c;
    for (int r = 0; r < NPG; ++r) s += base[(size_t)r * EMBD];
    g[gid * EMBD + c] = s * (1.0f / (float)NPG);
}

__global__ void head_kernel(const float* __restrict__ g, const float* __restrict__ W1,
                            const float* __restrict__ b1, const float* __restrict__ W2,
                            const float* __restrict__ b2, float* __restrict__ out) {
    __shared__ float gs[128];
    __shared__ float hs[128];
    int gid = blockIdx.x, c = threadIdx.x;           // blockDim = 128
    gs[c] = g[gid * EMBD + c];
    __syncthreads();
    float h = b1[c];
    for (int k = 0; k < 128; ++k) h += gs[k] * W1[k * 128 + c];
    hs[c] = fmaxf(h, 0.0f) * W2[c];
    __syncthreads();
    for (int s = 64; s > 0; s >>= 1) {
        if (c < s) hs[c] += hs[c + s];
        __syncthreads();
    }
    if (c == 0) out[gid] = hs[0] + b2[0];
}

// ---------------------------------------------------------------------------
extern "C" void kernel_launch(void* const* d_in, const int* in_sizes, int n_in,
                              void* d_out, int out_size, void* d_ws, size_t ws_size,
                              hipStream_t stream) {
    const float* pos   = (const float*)d_in[0];
    const float* emb   = (const float*)d_in[1];
    const float* Wf    = (const float*)d_in[2];
    const float* bfp   = (const float*)d_in[3];
    const float* Ws    = (const float*)d_in[4];
    const float* bsp   = (const float*)d_in[5];
    const float* bn1g  = (const float*)d_in[6];
    const float* bn1b  = (const float*)d_in[7];
    const float* bn2g  = (const float*)d_in[8];
    const float* bn2b  = (const float*)d_in[9];
    const float* W1    = (const float*)d_in[10];
    const float* b1    = (const float*)d_in[11];
    const float* W2    = (const float*)d_in[12];
    const float* b2    = (const float*)d_in[13];
    const int*   z     = (const int*)d_in[14];
    const int*   ei    = (const int*)d_in[16];
    float* out = (float*)d_out;

    // Carve workspace (256B aligned chunks); total ~145 MB (< 192 MB L2)
    char* base = (char*)d_ws;
    size_t off = 0;
    auto carve = [&](size_t bytes) -> char* {
        char* p = base + off;
        off += (bytes + 255) & ~(size_t)255;
        return p;
    };
    float*          xf    = (float*)carve((size_t)NN * EMBD * 4);          // 25.6 MB
    unsigned short* xb    = (unsigned short*)carve((size_t)NN * EMBD * 2); // 12.8 MB
    unsigned short* P16   = (unsigned short*)carve((size_t)NN * 512 * 2);  // 51.2 MB
    float*          agg   = (float*)carve((size_t)NN * EMBD * 4);          // 25.6 MB
    float*          outb  = (float*)carve((size_t)NN * EMBD * 4);          // 25.6 MB
    float*          dist  = (float*)carve((size_t)NE * 4);                 // 3.2 MB
    unsigned short* Wcat  = (unsigned short*)carve((size_t)NL * 512 * 128 * 2);
    float*          stats = (float*)carve(256 * 4);
    float*          gbuf  = (float*)carve((size_t)NG * EMBD * 4);
    (void)ws_size; (void)n_in; (void)in_sizes; (void)out_size; (void)pos;

    prep_w_kernel<<<(NL * 512 * 128 + 255) / 256, 256, 0, stream>>>(Wf, Ws, Wcat);
    embed_kernel<<<(NN * EMBD + 255) / 256, 256, 0, stream>>>(z, emb, xf, xb);
    dist_kernel<<<(NE + 255) / 256, 256, 0, stream>>>(ei, pos, dist);

    for (int l = 0; l < NL; ++l) {
        gemm_node_kernel<<<dim3(NN / 16, 4), 256, 0, stream>>>(
            xb, Wcat + (size_t)l * 512 * 128, P16);
        zero_aggstats_kernel<<<(NN * EMBD + 256 + 255) / 256, 256, 0, stream>>>(agg, stats);
        edge_msg_kernel<<<(NE * 32) / 256, 256, 0, stream>>>(
            ei, dist, P16, Wf, Ws, bfp, bsp, l, agg);
        bn_reduce_kernel<<<256, 128, 0, stream>>>(agg, stats);
        bn_apply1_kernel<<<(NN * EMBD + 255) / 256, 256, 0, stream>>>(
            agg, stats, bn1g, bn1b, xf, l, outb);
        zero_stats_kernel<<<1, 256, 0, stream>>>(stats);
        bn_reduce_kernel<<<256, 128, 0, stream>>>(outb, stats);
        bn_apply2_kernel<<<(NN * EMBD + 255) / 256, 256, 0, stream>>>(
            outb, stats, bn2g, bn2b, l, xf, xb);
    }

    pool_kernel<<<NG, 128, 0, stream>>>(xf, gbuf);
    head_kernel<<<NG, 128, 0, stream>>>(gbuf, W1, b1, W2, b2, out);
}
